// JointCrossAttention_67851893342910
// MI455X (gfx1250) — matrix operand order repaired and checked by reading, not compiled, not run on hardware
//
#include <hip/hip_runtime.h>
#include <hip/hip_bf16.h>
#include <math.h>

// ---------------- problem constants ----------------
#define BATCH 2
#define NI    2048
#define NJ    2048
#define DIMX  512
#define CDIMX 512
#define NH    8
#define DHEAD 64
#define INNER 512          // NH * DHEAD
#define SIMSCALE 0.125f    // DHEAD^-0.5
#define LNEPS 1e-5f

#define LSTR  72           // LDS row stride in bf16 elems (144B, 16B aligned)

typedef __attribute__((ext_vector_type(16))) __bf16          v16bf;
typedef __attribute__((ext_vector_type(8)))  float           v8f;
typedef __attribute__((ext_vector_type(8)))  unsigned short  ushort8;
typedef __attribute__((ext_vector_type(16))) unsigned short  ushort16;
typedef __attribute__((ext_vector_type(4)))  unsigned short  ushort4v;
typedef __attribute__((ext_vector_type(4)))  float           float4v;
typedef __attribute__((ext_vector_type(4)))  unsigned int    uint4v;
typedef __attribute__((ext_vector_type(4)))  int             int4v;

// --------- CDNA5 async global->LDS copy (bypasses VGPRs, uses ASYNCcnt) ----
#if defined(__gfx1250__) &&                                              \
    __has_builtin(__builtin_amdgcn_global_load_async_to_lds_b128) &&     \
    __has_builtin(__builtin_amdgcn_s_wait_asynccnt)
#define ASYNC_LDS 1
typedef __attribute__((address_space(1))) int4v* g_int4p;
typedef __attribute__((address_space(3))) int4v* l_int4p;
#else
#define ASYNC_LDS 0
#endif

__device__ __forceinline__ void cp_g2l_b128(const void* g, void* l) {
#if ASYNC_LDS
  __builtin_amdgcn_global_load_async_to_lds_b128(
      (g_int4p)(size_t)g, (l_int4p)(size_t)(__attribute__((address_space(3))) char*)l,
      0, 0);
#else
  *(uint4v*)l = *(const uint4v*)g;
#endif
}

__device__ __forceinline__ void async_fence() {
#if ASYNC_LDS
  __builtin_amdgcn_s_wait_asynccnt(0);
#endif
}

// f32 -> bf16 round-to-nearest-even
__device__ __forceinline__ unsigned short f2bf(float f) {
  unsigned int u = __builtin_bit_cast(unsigned int, f);
  u += 0x7FFFu + ((u >> 16) & 1u);
  return (unsigned short)(u >> 16);
}

// A fragment (16x32 bf16): lanes 0-15 row M=lane hold K {kh..kh+7, kh+16..kh+23}
// with kh=0; lanes 16-31 same rows, kh=8.  (ISA 7.12.2 16-bit A layout)
__device__ __forceinline__ v16bf frag_a(const unsigned short* tile, int lane) {
  int m  = lane & 15;
  int kh = (lane & 16) ? 8 : 0;
  ushort8 lo = *(const ushort8*)(tile + m * LSTR + kh);
  ushort8 hi = *(const ushort8*)(tile + m * LSTR + kh + 16);
  ushort16 c;
#pragma unroll
  for (int t = 0; t < 8; ++t) { c[t] = lo[t]; c[t + 8] = hi[t]; }
  return __builtin_bit_cast(v16bf, c);
}

// B fragment (32x16 bf16) from B^T tile stored [N][K]: lane n = lane&15 holds
// K {kb..kb+7, kb+16..kb+23}, kb=0 for lanes 0-15, kb=8 for lanes 16-31.
__device__ __forceinline__ v16bf frag_b(const unsigned short* tile, int lane) {
  int n  = lane & 15;
  int kb = (lane & 16) ? 8 : 0;
  ushort8 lo = *(const ushort8*)(tile + n * LSTR + kb);
  ushort8 hi = *(const ushort8*)(tile + n * LSTR + kb + 16);
  ushort16 c;
#pragma unroll
  for (int t = 0; t < 8; ++t) { c[t] = lo[t]; c[t + 8] = hi[t]; }
  return __builtin_bit_cast(v16bf, c);
}

__device__ __forceinline__ v8f wmma_bf16(v16bf a, v16bf b, v8f c) {
  return __builtin_amdgcn_wmma_f32_16x16x32_bf16(false, a, false, b,
                                                 (short)0, c, false, false);
}

// ---------------- layernorm + bf16 convert ----------------
__global__ __launch_bounds__(256) void ln_bf16_kernel(
    const float* __restrict__ x, const float* __restrict__ g,
    const float* __restrict__ be, unsigned short* __restrict__ out) {
  const int D = 512;
  int row = blockIdx.x, tid = threadIdx.x;
  const float* p = x + (size_t)row * D;
  float v0 = p[tid], v1 = p[tid + 256];
  float s = v0 + v1, ss = v0 * v0 + v1 * v1;
  __shared__ float r1[256], r2[256];
  r1[tid] = s; r2[tid] = ss;
  __syncthreads();
  for (int w = 128; w > 0; w >>= 1) {
    if (tid < w) { r1[tid] += r1[tid + w]; r2[tid] += r2[tid + w]; }
    __syncthreads();
  }
  float mean = r1[0] * (1.0f / D);
  float var  = r2[0] * (1.0f / D) - mean * mean;
  float rstd = rsqrtf(var + LNEPS);
  out[(size_t)row * D + tid]       = f2bf((v0 - mean) * rstd * g[tid] + be[tid]);
  out[(size_t)row * D + tid + 256] = f2bf((v1 - mean) * rstd * g[tid + 256] + be[tid + 256]);
}

// ---------------- weight transpose + bf16 convert: WT[n*K+k] = W[k*N+n] ----
__global__ __launch_bounds__(256) void wt_kernel(
    const float* __restrict__ W, unsigned short* __restrict__ WT, int K, int N) {
  int idx = blockIdx.x * 256 + threadIdx.x;
  if (idx >= K * N) return;
  int n = idx / K, k = idx - n * K;
  WT[idx] = f2bf(W[(size_t)k * N + n]);
}

// ---------------- generic bf16 WMMA GEMM: C[M][512] = A[M][K] * BT[512][K]^T
// mode 0: store bf16 split heads [b,h,row,d]
// mode 1: store bf16 split heads transposed [b,h,d,row]
// mode 2: store f32 + bias (final projection)
__global__ __launch_bounds__(256) void gemm_proj_kernel(
    const unsigned short* __restrict__ A, const unsigned short* __restrict__ BT,
    int K, int mode, unsigned short* __restrict__ ob16,
    float* __restrict__ of32, const float* __restrict__ bias) {
  __shared__ __align__(16) unsigned short As[128 * LSTR];
  __shared__ __align__(16) unsigned short Bs[64 * LSTR];
  int tid = threadIdx.x, lane = tid & 31, w = tid >> 5;
  int wm = w & 3, wn = w >> 2;
  int n0 = blockIdx.x * 64, m0 = blockIdx.y * 128;
  v8f acc[2][2] = {};
  for (int k0 = 0; k0 < K; k0 += 64) {
#pragma unroll
    for (int q = 0; q < 4; ++q) {
      int cid = q * 256 + tid;
      int r = cid >> 3, c = (cid & 7) * 8;
      cp_g2l_b128(A + (size_t)(m0 + r) * K + k0 + c, As + r * LSTR + c);
    }
#pragma unroll
    for (int q = 0; q < 2; ++q) {
      int cid = q * 256 + tid;
      int r = cid >> 3, c = (cid & 7) * 8;
      cp_g2l_b128(BT + (size_t)(n0 + r) * K + k0 + c, Bs + r * LSTR + c);
    }
    async_fence();
    __syncthreads();
#pragma unroll
    for (int kk = 0; kk < 64; kk += 32) {
      v16bf a0 = frag_a(As + (wm * 32) * LSTR + kk, lane);
      v16bf a1 = frag_a(As + (wm * 32 + 16) * LSTR + kk, lane);
      v16bf b0 = frag_b(Bs + (wn * 32) * LSTR + kk, lane);
      v16bf b1 = frag_b(Bs + (wn * 32 + 16) * LSTR + kk, lane);
      acc[0][0] = wmma_bf16(a0, b0, acc[0][0]);
      acc[0][1] = wmma_bf16(a0, b1, acc[0][1]);
      acc[1][0] = wmma_bf16(a1, b0, acc[1][0]);
      acc[1][1] = wmma_bf16(a1, b1, acc[1][1]);
    }
    __syncthreads();
  }
  int hi8 = (lane & 16) ? 8 : 0, nl = lane & 15;
#pragma unroll
  for (int i = 0; i < 2; ++i)
#pragma unroll
    for (int j = 0; j < 2; ++j) {
      int cg = n0 + wn * 32 + j * 16 + nl;
#pragma unroll
      for (int r = 0; r < 8; ++r) {
        int rowg = m0 + wm * 32 + i * 16 + hi8 + r;
        float v = acc[i][j][r];
        if (mode == 2) {
          of32[(size_t)rowg * INNER + cg] = v + bias[cg];
        } else {
          int b = rowg >> 11, irow = rowg & 2047;  // rows-per-batch = 2048
          int h = cg >> 6, d = cg & 63;
          size_t dst = (mode == 0)
              ? ((((size_t)b * NH + h) * NI + irow) * DHEAD + d)
              : ((((size_t)b * NH + h) * DHEAD + d) * NI + irow);
          ob16[dst] = f2bf(v);
        }
      }
    }
}

// ---------------- sim = qk @ cqk^T * SCALE  (per b,h; K=64) ----------------
__global__ __launch_bounds__(256) void sim_kernel(
    const unsigned short* __restrict__ qk, const unsigned short* __restrict__ cqk,
    float* __restrict__ sim) {
  __shared__ __align__(16) unsigned short As[128 * LSTR];
  __shared__ __align__(16) unsigned short Bs[64 * LSTR];
  int tid = threadIdx.x, lane = tid & 31, w = tid >> 5;
  int wm = w & 3, wn = w >> 2;
  int n0 = blockIdx.x * 64, m0 = blockIdx.y * 128, bh = blockIdx.z;
  const unsigned short* Ab = qk  + (size_t)bh * NI * DHEAD;
  const unsigned short* Bb = cqk + (size_t)bh * NJ * DHEAD;
#pragma unroll
  for (int q = 0; q < 4; ++q) {
    int cid = q * 256 + tid;
    int r = cid >> 3, c = (cid & 7) * 8;
    cp_g2l_b128(Ab + (size_t)(m0 + r) * DHEAD + c, As + r * LSTR + c);
  }
#pragma unroll
  for (int q = 0; q < 2; ++q) {
    int cid = q * 256 + tid;
    int r = cid >> 3, c = (cid & 7) * 8;
    cp_g2l_b128(Bb + (size_t)(n0 + r) * DHEAD + c, Bs + r * LSTR + c);
  }
  async_fence();
  __syncthreads();
  v8f acc[2][2] = {};
#pragma unroll
  for (int kk = 0; kk < 64; kk += 32) {
    v16bf a0 = frag_a(As + (wm * 32) * LSTR + kk, lane);
    v16bf a1 = frag_a(As + (wm * 32 + 16) * LSTR + kk, lane);
    v16bf b0 = frag_b(Bs + (wn * 32) * LSTR + kk, lane);
    v16bf b1 = frag_b(Bs + (wn * 32 + 16) * LSTR + kk, lane);
    acc[0][0] = wmma_bf16(a0, b0, acc[0][0]);
    acc[0][1] = wmma_bf16(a0, b1, acc[0][1]);
    acc[1][0] = wmma_bf16(a1, b0, acc[1][0]);
    acc[1][1] = wmma_bf16(a1, b1, acc[1][1]);
  }
  float* Sb = sim + (size_t)bh * NI * NJ;
  int hi8 = (lane & 16) ? 8 : 0, nl = lane & 15;
#pragma unroll
  for (int i = 0; i < 2; ++i)
#pragma unroll
    for (int j = 0; j < 2; ++j) {
      int cg = n0 + wn * 32 + j * 16 + nl;
#pragma unroll
      for (int r = 0; r < 8; ++r) {
        int rowg = m0 + wm * 32 + i * 16 + hi8 + r;
        Sb[(size_t)rowg * NJ + cg] = acc[i][j][r] * SIMSCALE;
      }
    }
}

// ---------------- softmax statistics (online max/sum) ----------------
__global__ __launch_bounds__(256) void row_stats_kernel(
    const float* __restrict__ sim, float* __restrict__ rmax, float* __restrict__ rinv) {
  size_t row = blockIdx.x;
  const float* p = sim + row * NJ;
  int tid = threadIdx.x;
  float m = -1e30f, l = 0.f;
  for (int t = tid; t < NJ; t += 256) {
    float x = p[t];
    float nm = fmaxf(m, x);
    l = l * __expf(m - nm) + __expf(x - nm);
    m = nm;
  }
  __shared__ float sm[256], sl[256];
  sm[tid] = m; sl[tid] = l;
  __syncthreads();
  for (int w = 128; w > 0; w >>= 1) {
    if (tid < w) {
      float m2 = sm[tid + w], l2 = sl[tid + w];
      float nm = fmaxf(sm[tid], m2);
      sl[tid] = sl[tid] * __expf(sm[tid] - nm) + l2 * __expf(m2 - nm);
      sm[tid] = nm;
    }
    __syncthreads();
  }
  if (tid == 0) { rmax[row] = sm[0]; rinv[row] = 1.0f / sl[0]; }
}

__global__ __launch_bounds__(256) void col_stats_kernel(
    const float* __restrict__ sim, float* __restrict__ cmax, float* __restrict__ cinv) {
  int bh = blockIdx.y;
  int j = blockIdx.x * 256 + threadIdx.x;
  const float* p = sim + (size_t)bh * NI * NJ + j;
  float m = -1e30f, l = 0.f;
  for (int i = 0; i < NI; ++i) {
    float x = p[(size_t)i * NJ];
    float nm = fmaxf(m, x);
    l = l * __expf(m - nm) + __expf(x - nm);
    m = nm;
  }
  cmax[(size_t)bh * NJ + j] = m;
  cinv[(size_t)bh * NJ + j] = 1.0f / l;
}

// ---------------- out = softmax_row(sim) @ cv  (M=I, N=64, K=J) ------------
__global__ __launch_bounds__(256) void attn_pv_kernel(
    const float* __restrict__ sim, const float* __restrict__ rmax,
    const float* __restrict__ rinv, const unsigned short* __restrict__ cvT,
    unsigned short* __restrict__ outm) {
  __shared__ __align__(16) unsigned short As[128 * LSTR];
  __shared__ __align__(16) unsigned short Bs[64 * LSTR];
  int tid = threadIdx.x, lane = tid & 31, w = tid >> 5;
  int wm = w & 3, wn = w >> 2;
  int bh = blockIdx.y, m0 = blockIdx.x * 128;
  int b = bh / NH, h = bh % NH;
  const float* Sb = sim + (size_t)bh * NI * NJ;
  const float* rm = rmax + (size_t)bh * NI;
  const float* ri = rinv + (size_t)bh * NI;
  const unsigned short* Bb = cvT + (size_t)bh * DHEAD * NJ;
  v8f acc[2][2] = {};
  for (int k0 = 0; k0 < NJ; k0 += 64) {
    // B tile: async direct-to-LDS
#pragma unroll
    for (int q = 0; q < 2; ++q) {
      int cid = q * 256 + tid;
      int r = cid >> 3, c = (cid & 7) * 8;
      cp_g2l_b128(Bb + (size_t)r * NJ + k0 + c, Bs + r * LSTR + c);
    }
    // A tile: on-the-fly softmax(sim) -> bf16 (must go through VGPRs)
#pragma unroll
    for (int q = 0; q < 8; ++q) {
      int cid = q * 256 + tid;
      int r = cid >> 4, c = (cid & 15) * 4;
      float4v s = *(const float4v*)(Sb + (size_t)(m0 + r) * NJ + k0 + c);
      float mr = rm[m0 + r], vr = ri[m0 + r];
      ushort4v o;
#pragma unroll
      for (int t = 0; t < 4; ++t) o[t] = f2bf(__expf(s[t] - mr) * vr);
      *(ushort4v*)(As + r * LSTR + c) = o;
    }
    async_fence();
    __syncthreads();
#pragma unroll
    for (int kk = 0; kk < 64; kk += 32) {
      v16bf a0 = frag_a(As + (wm * 32) * LSTR + kk, lane);
      v16bf a1 = frag_a(As + (wm * 32 + 16) * LSTR + kk, lane);
      v16bf b0 = frag_b(Bs + (wn * 32) * LSTR + kk, lane);
      v16bf b1 = frag_b(Bs + (wn * 32 + 16) * LSTR + kk, lane);
      acc[0][0] = wmma_bf16(a0, b0, acc[0][0]);
      acc[0][1] = wmma_bf16(a0, b1, acc[0][1]);
      acc[1][0] = wmma_bf16(a1, b0, acc[1][0]);
      acc[1][1] = wmma_bf16(a1, b1, acc[1][1]);
    }
    __syncthreads();
  }
  int hi8 = (lane & 16) ? 8 : 0, nl = lane & 15;
#pragma unroll
  for (int i = 0; i < 2; ++i)
#pragma unroll
    for (int j = 0; j < 2; ++j) {
      int d = wn * 32 + j * 16 + nl;
#pragma unroll
      for (int r = 0; r < 8; ++r) {
        int irow = m0 + wm * 32 + i * 16 + hi8 + r;
        outm[((size_t)b * NI + irow) * INNER + h * DHEAD + d] = f2bf(acc[i][j][r]);
      }
    }
}

// ---------------- cout = softmax_col(sim)^T @ v  (M=J, N=64, K=I) ----------
__global__ __launch_bounds__(256) void cattn_pv_kernel(
    const float* __restrict__ sim, const float* __restrict__ cmax,
    const float* __restrict__ cinv, const unsigned short* __restrict__ vT,
    unsigned short* __restrict__ coutm) {
  __shared__ __align__(16) unsigned short As[128 * LSTR];
  __shared__ __align__(16) unsigned short Bs[64 * LSTR];
  int tid = threadIdx.x, lane = tid & 31, w = tid >> 5;
  int wm = w & 3, wn = w >> 2;
  int bh = blockIdx.y, m0 = blockIdx.x * 128;  // m over j
  int b = bh / NH, h = bh % NH;
  const float* Sb = sim + (size_t)bh * NI * NJ;
  const float* cm = cmax + (size_t)bh * NJ;
  const float* ci = cinv + (size_t)bh * NJ;
  const unsigned short* Bb = vT + (size_t)bh * DHEAD * NI;
  v8f acc[2][2] = {};
  for (int k0 = 0; k0 < NI; k0 += 64) {
    // B tile: async direct-to-LDS
#pragma unroll
    for (int q = 0; q < 2; ++q) {
      int cid = q * 256 + tid;
      int r = cid >> 3, c = (cid & 7) * 8;
      cp_g2l_b128(Bb + (size_t)r * NI + k0 + c, Bs + r * LSTR + c);
    }
    // A tile: read sim[k0..k0+63][m0..m0+127], transpose+exp into As[j'][i']
#pragma unroll
    for (int q = 0; q < 8; ++q) {
      int cid = q * 256 + tid;
      int ir = cid >> 5, jc = (cid & 31) * 4;
      float4v s = *(const float4v*)(Sb + (size_t)(k0 + ir) * NJ + m0 + jc);
#pragma unroll
      for (int t = 0; t < 4; ++t) {
        int jg = m0 + jc + t;
        As[(jc + t) * LSTR + ir] = f2bf(__expf(s[t] - cm[jg]) * ci[jg]);
      }
    }
    async_fence();
    __syncthreads();
#pragma unroll
    for (int kk = 0; kk < 64; kk += 32) {
      v16bf a0 = frag_a(As + (wm * 32) * LSTR + kk, lane);
      v16bf a1 = frag_a(As + (wm * 32 + 16) * LSTR + kk, lane);
      v16bf b0 = frag_b(Bs + (wn * 32) * LSTR + kk, lane);
      v16bf b1 = frag_b(Bs + (wn * 32 + 16) * LSTR + kk, lane);
      acc[0][0] = wmma_bf16(a0, b0, acc[0][0]);
      acc[0][1] = wmma_bf16(a0, b1, acc[0][1]);
      acc[1][0] = wmma_bf16(a1, b0, acc[1][0]);
      acc[1][1] = wmma_bf16(a1, b1, acc[1][1]);
    }
    __syncthreads();
  }
  int hi8 = (lane & 16) ? 8 : 0, nl = lane & 15;
#pragma unroll
  for (int i = 0; i < 2; ++i)
#pragma unroll
    for (int j = 0; j < 2; ++j) {
      int d = wn * 32 + j * 16 + nl;
#pragma unroll
      for (int r = 0; r < 8; ++r) {
        int jrow = m0 + wm * 32 + i * 16 + hi8 + r;
        coutm[((size_t)b * NJ + jrow) * INNER + h * DHEAD + d] = f2bf(acc[i][j][r]);
      }
    }
}

// ---------------- host orchestration ----------------
extern "C" void kernel_launch(void* const* d_in, const int* in_sizes, int n_in,
                              void* d_out, int out_size, void* d_ws, size_t ws_size,
                              hipStream_t stream) {
  (void)in_sizes; (void)n_in; (void)out_size; (void)ws_size;
  const float* x      = (const float*)d_in[0];
  const float* ctx    = (const float*)d_in[1];
  const float* ln_g   = (const float*)d_in[2];
  const float* ln_b   = (const float*)d_in[3];
  const float* cln_g  = (const float*)d_in[4];
  const float* cln_b  = (const float*)d_in[5];
  const float* W_qk   = (const float*)d_in[6];
  const float* W_cqk  = (const float*)d_in[7];
  const float* W_v    = (const float*)d_in[8];
  const float* W_cv   = (const float*)d_in[9];
  const float* W_out  = (const float*)d_in[10];
  const float* b_out  = (const float*)d_in[11];
  const float* W_cout = (const float*)d_in[12];
  const float* b_cout = (const float*)d_in[13];
  float* out  = (float*)d_out;
  float* cout = out + (size_t)BATCH * NI * DIMX;

  char* ws = (char*)d_ws;
  size_t off = 0;
  auto carve = [&](size_t bytes) -> void* {
    void* p = ws + off;
    off += (bytes + 255) & ~(size_t)255;
    return p;
  };
  unsigned short* xnb   = (unsigned short*)carve((size_t)BATCH * NI * DIMX * 2);
  unsigned short* cnb   = (unsigned short*)carve((size_t)BATCH * NJ * CDIMX * 2);
  unsigned short* WqkT  = (unsigned short*)carve((size_t)DIMX * INNER * 2);
  unsigned short* WcqkT = (unsigned short*)carve((size_t)CDIMX * INNER * 2);
  unsigned short* WvT   = (unsigned short*)carve((size_t)DIMX * INNER * 2);
  unsigned short* WcvT  = (unsigned short*)carve((size_t)CDIMX * INNER * 2);
  unsigned short* WoutT = (unsigned short*)carve((size_t)INNER * DIMX * 2);
  unsigned short* WcoutT= (unsigned short*)carve((size_t)INNER * CDIMX * 2);
  unsigned short* qk    = (unsigned short*)carve((size_t)BATCH * NH * NI * DHEAD * 2);
  unsigned short* cqk   = (unsigned short*)carve((size_t)BATCH * NH * NJ * DHEAD * 2);
  unsigned short* vT    = (unsigned short*)carve((size_t)BATCH * NH * DHEAD * NI * 2);
  unsigned short* cvT   = (unsigned short*)carve((size_t)BATCH * NH * DHEAD * NJ * 2);
  unsigned short* attnm = (unsigned short*)carve((size_t)BATCH * NI * INNER * 2);
  unsigned short* coutm = (unsigned short*)carve((size_t)BATCH * NJ * INNER * 2);
  float* rmax = (float*)carve((size_t)BATCH * NH * NI * 4);
  float* rinv = (float*)carve((size_t)BATCH * NH * NI * 4);
  float* cmax = (float*)carve((size_t)BATCH * NH * NJ * 4);
  float* cinv = (float*)carve((size_t)BATCH * NH * NJ * 4);
  float* sim  = (float*)carve((size_t)BATCH * NH * NI * NJ * 4);

  dim3 blk(256);
  int wgrid = (512 * 512 + 255) / 256;
  wt_kernel<<<wgrid, blk, 0, stream>>>(W_qk,   WqkT,   DIMX,  INNER);
  wt_kernel<<<wgrid, blk, 0, stream>>>(W_cqk,  WcqkT,  CDIMX, INNER);
  wt_kernel<<<wgrid, blk, 0, stream>>>(W_v,    WvT,    DIMX,  INNER);
  wt_kernel<<<wgrid, blk, 0, stream>>>(W_cv,   WcvT,   CDIMX, INNER);
  wt_kernel<<<wgrid, blk, 0, stream>>>(W_out,  WoutT,  INNER, DIMX);
  wt_kernel<<<wgrid, blk, 0, stream>>>(W_cout, WcoutT, INNER, CDIMX);

  ln_bf16_kernel<<<BATCH * NI, blk, 0, stream>>>(x,   ln_g,  ln_b,  xnb);
  ln_bf16_kernel<<<BATCH * NJ, blk, 0, stream>>>(ctx, cln_g, cln_b, cnb);

  dim3 gp(INNER / 64, (BATCH * NI) / 128);
  gemm_proj_kernel<<<gp, blk, 0, stream>>>(xnb, WqkT,  DIMX,  0, qk,  nullptr, nullptr);
  gemm_proj_kernel<<<gp, blk, 0, stream>>>(xnb, WvT,   DIMX,  1, vT,  nullptr, nullptr);
  gemm_proj_kernel<<<gp, blk, 0, stream>>>(cnb, WcqkT, CDIMX, 0, cqk, nullptr, nullptr);
  gemm_proj_kernel<<<gp, blk, 0, stream>>>(cnb, WcvT,  CDIMX, 1, cvT, nullptr, nullptr);

  dim3 gs(NJ / 64, NI / 128, BATCH * NH);
  sim_kernel<<<gs, blk, 0, stream>>>(qk, cqk, sim);

  row_stats_kernel<<<BATCH * NH * NI, blk, 0, stream>>>(sim, rmax, rinv);
  dim3 gc(NJ / 256, BATCH * NH);
  col_stats_kernel<<<gc, blk, 0, stream>>>(sim, cmax, cinv);

  dim3 ga(NI / 128, BATCH * NH);
  attn_pv_kernel<<<ga, blk, 0, stream>>>(sim, rmax, rinv, cvT, attnm);
  dim3 gb(NJ / 128, BATCH * NH);
  cattn_pv_kernel<<<gb, blk, 0, stream>>>(sim, cmax, cinv, vT, coutm);

  gemm_proj_kernel<<<gp, blk, 0, stream>>>(attnm, WoutT,  INNER, 2, nullptr, out,  b_out);
  gemm_proj_kernel<<<gp, blk, 0, stream>>>(coutm, WcoutT, INNER, 2, nullptr, cout, b_cout);
}
